// BiLSTM_CRF_54528904790760
// MI455X (gfx1250) — matrix operand order repaired and checked by reading, not compiled, not run on hardware
//
#include <hip/hip_runtime.h>
#include <hip/hip_bf16.h>

// ---------------------------------------------------------------------------
// BiLSTM-CRF for MI455X (gfx1250, wave32, WMMA).
//   V=32000 E=256 HD=512 T=9 B=64 S=512 H=256, 4H=1024
// Strategy:
//   1) embed -> x (f16, time-major [S,B,E])
//   2) big WMMA GEMM: xg[dir] = x @ Wih.T + b   (off the serial path)
//   3) per-direction sequential scan (1 workgroup each, grid=2):
//      g = h @ Whh.T via v_wmma_f32_16x16x32_f16 (h in LDS f16, g in LDS f32,
//      c in registers), fused sigmoid/tanh gate update, masked freeze.
//   4) feats = hs @ W_out.T + b_out (T=9, scalar)
//   5) Viterbi forward + backtrace (B=64 threads)
// ---------------------------------------------------------------------------

typedef __attribute__((ext_vector_type(16))) _Float16 v16h;
typedef __attribute__((ext_vector_type(8)))  _Float16 v8h;
typedef __attribute__((ext_vector_type(8)))  float    v8f;

#define E_   256
#define HD_  512
#define T_   9
#define B_   64
#define S_   512
#define H_   256
#define G4_  1024           // 4*H
#define NEG_ (-10000.0f)
#define HLD  264            // padded LDS row stride for h (halves)
#define SMEM_BYTES (B_ * HLD * 2 + B_ * G4_ * 4)   // 33792 + 262144 = 295936 B

// ---- WMMA fragment helpers (layouts per CDNA5 ISA 7.12.2) ------------------

// A-matrix 16x32 f16 from row-major src (ld halves between rows).
// lane<16 : halves[0..7]=K0..7,  halves[8..15]=K16..23  (row M = lane)
// lane>=16: halves[0..7]=K8..15, halves[8..15]=K24..31  (row M = lane-16)
__device__ __forceinline__ v16h load_a_frag(const _Float16* src, int ld) {
  int lane = threadIdx.x & 31;
  int row  = lane & 15;
  int kb   = (lane & 16) ? 8 : 0;
  const _Float16* p = src + row * ld + kb;
  union { v16h v; v8h h[2]; } u;
  u.h[0] = *(const v8h*)(p);
  u.h[1] = *(const v8h*)(p + 16);
  return u.v;
}

// B-matrix 32x16 f16, B[k][n] = W[n][k] with W row-major [N,K] (ld halves).
// lane<16 : column N=lane,   halves[0..7]=K0..7,  halves[8..15]=K16..23
// lane>=16: column N=lane-16, halves[0..7]=K8..15, halves[8..15]=K24..31
__device__ __forceinline__ v16h load_b_frag(const _Float16* w, int ld) {
  int lane = threadIdx.x & 31;
  int n    = lane & 15;
  int kb   = (lane & 16) ? 8 : 0;
  const _Float16* p = w + n * ld + kb;
  union { v16h v; v8h h[2]; } u;
  u.h[0] = *(const v8h*)(p);
  u.h[1] = *(const v8h*)(p + 16);
  return u.v;
}

__device__ __forceinline__ v8f wmma_f16(v16h a, v16h b, v8f c) {
  return __builtin_amdgcn_wmma_f32_16x16x32_f16(false, a, false, b,
                                                (short)0, c, false, false);
}

__device__ __forceinline__ float sigmoidf_(float x) {
  return 1.0f / (1.0f + expf(-x));
}

// ---- 1) embedding lookup + f32->f16, time-major --------------------------
__global__ void embed_kernel(const int* __restrict__ ids,
                             const float* __restrict__ table,
                             _Float16* __restrict__ xh) {
  int idx = blockIdx.x * blockDim.x + threadIdx.x;     // S*B*E threads
  int e  = idx & (E_ - 1);
  int sb = idx >> 8;                                   // s*B + b
  int b  = sb & (B_ - 1);
  int s  = sb >> 6;
  int id = ids[b * S_ + s];
  xh[idx] = (_Float16)table[id * E_ + e];
}

// ---- f32 -> f16 weight cast ------------------------------------------------
__global__ void cast_kernel(const float* __restrict__ src,
                            _Float16* __restrict__ dst, int n) {
  int i = blockIdx.x * blockDim.x + threadIdx.x;
  if (i < n) dst[i] = (_Float16)src[i];
}

// ---- 2) xg = x @ Wih.T + b  (M=S*B=32768, K=256, N=1024) -------------------
// 256 threads = 8 waves/block; each wave does a 16(M) x 256(N) strip.
__global__ void pregemm_kernel(const _Float16* __restrict__ x,
                               const _Float16* __restrict__ w,
                               const float* __restrict__ bias,
                               float* __restrict__ out) {
  int wave = threadIdx.x >> 5;
  int task = blockIdx.x * 8 + wave;      // 8192 tasks = 2048 Mtiles * 4 strips
  int mt = task >> 2;
  int ns = task & 3;
  const _Float16* abase = x + (size_t)mt * 16 * E_;
  v16h af[8];
#pragma unroll
  for (int k = 0; k < 8; ++k) af[k] = load_a_frag(abase + k * 32, E_);
  int lane = threadIdx.x & 31;
  int n    = lane & 15;
  int mb   = (lane & 16) ? 8 : 0;
  for (int t = 0; t < 16; ++t) {
    int n0 = ns * 256 + t * 16;
    v8f acc = {};
#pragma unroll
    for (int k = 0; k < 8; ++k) {
      v16h bf = load_b_frag(w + (size_t)n0 * E_ + k * 32, E_);
      acc = wmma_f16(af[k], bf, acc);
    }
    float bv = bias[n0 + n];
    float* dst = out + (size_t)mt * 16 * G4_ + n0;
#pragma unroll
    for (int i = 0; i < 8; ++i) dst[(mb + i) * G4_ + n] = acc[i] + bv;
  }
}

// ---- 3) sequential bidirectional LSTM scan (grid=2: dir 0=fwd, 1=rev) -----
__global__ void lstm_scan_kernel(const int* __restrict__ ids,
                                 const _Float16* __restrict__ whh_f,
                                 const _Float16* __restrict__ whh_r,
                                 const float* __restrict__ xg_f,
                                 const float* __restrict__ xg_r,
                                 float* __restrict__ hs) {
  extern __shared__ __align__(16) char smem[];
  _Float16* h_sh = (_Float16*)smem;                    // [64][HLD] f16
  float*    g_sh = (float*)(smem + B_ * HLD * 2);      // [64][1024] f32

  const int dir = blockIdx.x;
  const _Float16* whh = dir ? whh_r : whh_f;
  const float*    xg  = dir ? xg_r  : xg_f;
  const int tid = threadIdx.x;                         // 512 threads, 16 waves

  for (int i = tid; i < B_ * HLD; i += 512) h_sh[i] = (_Float16)0.0f;
  float c[32];
#pragma unroll
  for (int j = 0; j < 32; ++j) c[j] = 0.0f;

  const int wave   = tid >> 5;
  const int mt     = wave & 3;          // batch tile (4 x 16 rows)
  const int nsbase = (wave >> 2) * 16;  // 16 N-tiles per wave
  const int lane   = tid & 31;
  const int cn_    = lane & 15;
  const int cmb    = (lane & 16) ? 8 : 0;
  const int eb     = tid >> 3;          // elementwise: batch row
  const int en0    = (tid & 7) * 32;    // elementwise: column base
  __syncthreads();

  for (int si = 0; si < S_; ++si) {
    int s = dir ? (S_ - 1 - si) : si;

    // --- g_rec = h @ Whh.T  (M=64, K=256, N=1024) ---
    v16h af[8];
#pragma unroll
    for (int k = 0; k < 8; ++k)
      af[k] = load_a_frag(h_sh + mt * 16 * HLD + k * 32, HLD);
    for (int t = 0; t < 16; ++t) {
      int n0 = (nsbase + t) * 16;
      v8f acc = {};
#pragma unroll
      for (int k = 0; k < 8; ++k) {
        v16h bf = load_b_frag(whh + (size_t)n0 * H_ + k * 32, H_);
        acc = wmma_f16(af[k], bf, acc);
      }
      float* dst = g_sh + (mt * 16) * G4_ + n0;
#pragma unroll
      for (int i = 0; i < 8; ++i) dst[(cmb + i) * G4_ + cn_] = acc[i];
    }
    __syncthreads();

    // --- gates + state update (mask derived from ids: 0 == padded) ---
    bool m = ids[eb * S_ + s] != 0;
    const float* xgrow = xg + (size_t)(s * B_ + eb) * G4_;
    float*       hsrow = hs + (size_t)(s * B_ + eb) * HD_ + dir * H_;
    const float* grow  = g_sh + eb * G4_;
#pragma unroll 4
    for (int j = 0; j < 32; ++j) {
      int n = en0 + j;
      float gi = grow[n]       + xgrow[n];
      float gf = grow[256 + n] + xgrow[256 + n];
      float gg = grow[512 + n] + xgrow[512 + n];
      float go = grow[768 + n] + xgrow[768 + n];
      float hn = 0.0f;
      if (m) {
        float cnv = sigmoidf_(gf) * c[j] + sigmoidf_(gi) * tanhf(gg);
        hn = sigmoidf_(go) * tanhf(cnv);
        c[j] = cnv;
        h_sh[eb * HLD + n] = (_Float16)hn;   // frozen when masked
      }
      hsrow[n] = hn;                          // hs is zeroed at masked steps
    }
    __syncthreads();
  }
}

// ---- 4) feats = hs @ W_out.T + b_out (T=9) --------------------------------
__global__ void feats_kernel(const float* __restrict__ hs,
                             const float* __restrict__ wout,
                             const float* __restrict__ bout,
                             float* __restrict__ feats) {
  int idx = blockIdx.x * blockDim.x + threadIdx.x;
  if (idx >= S_ * B_ * T_) return;
  int t  = idx % T_;
  int sb = idx / T_;
  const float* hrow = hs + (size_t)sb * HD_;
  const float* wrow = wout + t * HD_;
  float acc = bout[t];
#pragma unroll 8
  for (int k = 0; k < HD_; ++k) acc += hrow[k] * wrow[k];
  feats[idx] = acc;
}

// ---- 5) Viterbi decode (one thread per batch element) ---------------------
__global__ void viterbi_kernel(const int* __restrict__ ids,
                               const float* __restrict__ trans,
                               const float* __restrict__ feats,
                               unsigned char* __restrict__ bp,
                               int* __restrict__ out) {
  __shared__ float tr[T_ * T_];
  if (threadIdx.x < T_ * T_) tr[threadIdx.x] = trans[threadIdx.x];
  __syncthreads();
  int b = threadIdx.x;
  if (b >= B_) return;

  float v[T_];
#pragma unroll
  for (int t = 0; t < T_; ++t) v[t] = (t == 0) ? 0.0f : NEG_;

  for (int s = 0; s < S_; ++s) {
    bool m = ids[b * S_ + s] != 0;
    const float* f = feats + (size_t)(s * B_ + b) * T_;
    unsigned char* bprow = bp + (size_t)(s * B_ + b) * T_;
    float nv[T_];
#pragma unroll
    for (int cur = 0; cur < T_; ++cur) {
      float best = v[0] + tr[cur];
      int arg = 0;
#pragma unroll
      for (int prev = 1; prev < T_; ++prev) {
        float sc = v[prev] + tr[prev * T_ + cur];
        if (sc > best) { best = sc; arg = prev; }   // first-max like argmax
      }
      bprow[cur] = (unsigned char)arg;
      nv[cur] = best + f[cur];
    }
    if (m) {
#pragma unroll
      for (int t = 0; t < T_; ++t) v[t] = nv[t];
    }
  }

  // last = argmax_t(v[t] + transitions[PAD=0][t])
  float best = v[0] + tr[0];
  int last = 0;
#pragma unroll
  for (int t = 1; t < T_; ++t) {
    float sc = v[t] + tr[t];
    if (sc > best) { best = sc; last = t; }
  }
  out[b * S_ + S_ - 1] = last;
  int tag = last;
  for (int s = S_ - 1; s >= 1; --s) {
    int prev = (ids[b * S_ + s] != 0)
                   ? (int)bp[(size_t)(s * B_ + b) * T_ + tag] : 0;
    out[b * S_ + s - 1] = prev;
    tag = prev;
  }
}

// ---- workspace layout (bytes, 256-aligned regions) ------------------------
#define WS_XH     ((size_t)0)            // S*B*E f16          = 16 MiB
#define WS_WIHF   ((size_t)16777216)     // 1024*256 f16       = 512 KiB
#define WS_WHHF   ((size_t)17301504)
#define WS_WIHR   ((size_t)17825792)
#define WS_WHHR   ((size_t)18350080)
#define WS_XGF    ((size_t)18874368)     // S*B*1024 f32       = 128 MiB
#define WS_XGR    ((size_t)153092096)
#define WS_HS     ((size_t)287309824)    // S*B*512 f32        = 64 MiB
#define WS_FEATS  ((size_t)354418688)    // S*B*9 f32
#define WS_BP     ((size_t)355598336)    // S*B*9 u8

extern "C" void kernel_launch(void* const* d_in, const int* in_sizes, int n_in,
                              void* d_out, int out_size, void* d_ws, size_t ws_size,
                              hipStream_t stream) {
  (void)in_sizes; (void)n_in; (void)out_size; (void)ws_size;

  const int*   ids   = (const int*)d_in[0];
  // d_in[1] = attention_mask (unused; mask == (input_ids != 0) by construction)
  const float* table = (const float*)d_in[2];
  const float* Wih_f = (const float*)d_in[3];
  const float* Whh_f = (const float*)d_in[4];
  const float* b_f   = (const float*)d_in[5];
  const float* Wih_r = (const float*)d_in[6];
  const float* Whh_r = (const float*)d_in[7];
  const float* b_r   = (const float*)d_in[8];
  const float* W_out = (const float*)d_in[9];
  const float* b_out = (const float*)d_in[10];
  const float* trans = (const float*)d_in[11];

  char* ws = (char*)d_ws;
  _Float16* xh     = (_Float16*)(ws + WS_XH);
  _Float16* wihf_h = (_Float16*)(ws + WS_WIHF);
  _Float16* whhf_h = (_Float16*)(ws + WS_WHHF);
  _Float16* wihr_h = (_Float16*)(ws + WS_WIHR);
  _Float16* whhr_h = (_Float16*)(ws + WS_WHHR);
  float* xg_f  = (float*)(ws + WS_XGF);
  float* xg_r  = (float*)(ws + WS_XGR);
  float* hs    = (float*)(ws + WS_HS);
  float* feats = (float*)(ws + WS_FEATS);
  unsigned char* bp = (unsigned char*)(ws + WS_BP);

  // allow 289 KB dynamic LDS on the 320 KB WGP (host-side, capture-safe)
  (void)hipFuncSetAttribute(reinterpret_cast<const void*>(lstm_scan_kernel),
                            hipFuncAttributeMaxDynamicSharedMemorySize,
                            SMEM_BYTES);

  embed_kernel<<<(S_ * B_ * E_) / 256, 256, 0, stream>>>(ids, table, xh);
  cast_kernel<<<(G4_ * E_) / 256, 256, 0, stream>>>(Wih_f, wihf_h, G4_ * E_);
  cast_kernel<<<(G4_ * H_) / 256, 256, 0, stream>>>(Whh_f, whhf_h, G4_ * H_);
  cast_kernel<<<(G4_ * E_) / 256, 256, 0, stream>>>(Wih_r, wihr_h, G4_ * E_);
  cast_kernel<<<(G4_ * H_) / 256, 256, 0, stream>>>(Whh_r, whhr_h, G4_ * H_);

  pregemm_kernel<<<1024, 256, 0, stream>>>(xh, wihf_h, b_f, xg_f);
  pregemm_kernel<<<1024, 256, 0, stream>>>(xh, wihr_h, b_r, xg_r);

  lstm_scan_kernel<<<2, 512, SMEM_BYTES, stream>>>(ids, whhf_h, whhr_h,
                                                   xg_f, xg_r, hs);

  feats_kernel<<<(S_ * B_ * T_ + 255) / 256, 256, 0, stream>>>(hs, W_out,
                                                               b_out, feats);
  viterbi_kernel<<<1, 128, 0, stream>>>(ids, trans, feats, bp, (int*)d_out);
}
